// SelfAttention_65360812311069
// MI455X (gfx1250) — compile-verified, matrix operand order checked by reading, not run on hardware
//
#include <hip/hip_runtime.h>
#include <hip/hip_bf16.h>

// ---------------------------------------------------------------------------
// MI455X (gfx1250) fused self-attention:
//   prep  : fp32 weights -> bf16, transposed (so WMMA B-frags are contiguous);
//           g-weights left unscaled; log2(e) is applied at the proj store
//   proj  : f = x@wf+bf, g = (x@wg+bg)*log2e (row-major bf16), h = x@wh+bh
//           stored TRANSPOSED hT[C][N] bf16 (contiguous attention V B-frags)
//   attn  : flash-attention, 64-key chunks, bf16 WMMA for S = g f^T and
//           O = P V; online softmax in fp32: single v_exp_f32 per score,
//           row-sums accumulated by WMMA against a ones-matrix (co-executes
//           with VALU), max-rescale skipped by wave-uniform __any vote.
// ---------------------------------------------------------------------------

typedef __bf16 bf16_t;
typedef bf16_t v16bf __attribute__((ext_vector_type(16)));
typedef float  v8f   __attribute__((ext_vector_type(8)));

union Frag16 {
    v16bf          bf;
    uint4          u4[2];
    unsigned short us[16];
};

__device__ __forceinline__ unsigned short f2bf(float f) {
    __bf16 h = (__bf16)f;                    // native v_cvt_pk_bf16_f32
    return __builtin_bit_cast(unsigned short, h);
}

#if __has_builtin(__builtin_amdgcn_exp2f)
#define EXP2F(x) __builtin_amdgcn_exp2f(x)   // v_exp_f32 is natively 2^x
#else
#define EXP2F(x) __expf((x) * 0.6931471805599453f)
#endif
#define LOG2E 1.4426950408889634f

__device__ __forceinline__ v8f wmma_bf16(const Frag16& A, const Frag16& B, v8f C) {
    return __builtin_amdgcn_wmma_f32_16x16x32_bf16(
        /*neg_a=*/false, A.bf, /*neg_b=*/false, B.bf,
        /*c_mod=*/(short)0, C, /*reuse_a=*/false, /*reuse_b=*/false);
}

__device__ __forceinline__ Frag16 load_frag(const unsigned short* p) {
    // 16 bf16 per lane: halves [K+0..7] then [K+16..23] (caller pre-offsets by
    // base = 8*(lane>>4)); both chunks 16B aligned.
    Frag16 r;
    r.u4[0] = *(const uint4*)(p);
    r.u4[1] = *(const uint4*)(p + 16);
    return r;
}

// xor-butterfly within 16-lane rows via v_permlane16_b32 (VALU, not DS).
__device__ __forceinline__ float permxor(float v, unsigned lo, unsigned hi) {
    unsigned u = __builtin_bit_cast(unsigned, v);
    unsigned r = __builtin_amdgcn_permlane16(u, u, lo, hi, false, false);
    return __builtin_bit_cast(float, r);
}
#define XOR1_LO 0x67452301u
#define XOR1_HI 0xEFCDAB89u
#define XOR2_LO 0x54761032u
#define XOR2_HI 0xDCFE98BAu
#define XOR4_LO 0x32107654u
#define XOR4_HI 0xBA98FEDCu
#define XOR8_LO 0xFEDCBA98u
#define XOR8_HI 0x76543210u

// --------------------------- constants -------------------------------------
#define NB   8
#define NTOK 4096          // H*W per batch
#define CCH  256
#define CK   32

// ws element offsets (unsigned short elements)
#define OFF_WFT 0u
#define OFF_WGT 8192u
#define OFF_WHT 16384u
#define OFF_F   81920u                        // [B*N][32]
#define OFF_G   (81920u + 1048576u)           // [B*N][32]
#define OFF_HT  (81920u + 2097152u)           // [B][256][4096]

// ---------------------------------------------------------------------------
// Kernel 0: weight convert + transpose.  wfT/wgT: [32][256], whT: [256][256]
// ---------------------------------------------------------------------------
__global__ __launch_bounds__(256) void prep_kernel(
    const float* __restrict__ wf, const float* __restrict__ wg,
    const float* __restrict__ wh, unsigned short* __restrict__ ws)
{
    int id = blockIdx.x * 256 + threadIdx.x;
    if (id < 8192) {
        int n = id >> 8, k = id & 255;
        ws[OFF_WFT + id] = f2bf(wf[k * 32 + n]);
    } else if (id < 16384) {
        int j = id - 8192, n = j >> 8, k = j & 255;
        ws[OFF_WGT + j] = f2bf(wg[k * 32 + n]);   // log2e folded at proj store
    } else if (id < 81920) {
        int j = id - 16384, n = j >> 8, k = j & 255;
        ws[OFF_WHT + j] = f2bf(wh[k * 256 + n]);
    }
}

// ---------------------------------------------------------------------------
// Kernel 1: projections. One block = 16 tokens, 4 waves; wave w owns output
// tiles {w, w+4, w+8, w+12, w+16} of 20 (f:2, g:2, h:16), K=256 in 8 steps.
// ---------------------------------------------------------------------------
__global__ __launch_bounds__(128) void proj_kernel(
    const float* __restrict__ x, const unsigned short* __restrict__ ws_ro,
    const float* __restrict__ bfv, const float* __restrict__ bgv,
    const float* __restrict__ bhv, unsigned short* __restrict__ ws)
{
    const unsigned short* wfT = ws_ro + OFF_WFT;
    const unsigned short* wgT = ws_ro + OFF_WGT;
    const unsigned short* whT = ws_ro + OFF_WHT;
    unsigned short* fo  = ws + OFF_F;
    unsigned short* go  = ws + OFF_G;
    unsigned short* hTo = ws + OFF_HT;

    const int tid = threadIdx.x, wave = tid >> 5, lane = tid & 31;
    const int lh = lane & 15, hh = lane >> 4, base = hh * 8;
    const int t0 = blockIdx.x * 16;            // global token tile
    const int bb = t0 >> 12, n0 = t0 & 4095;   // batch / token-in-batch

    v8f acc[5];
#pragma unroll
    for (int i = 0; i < 5; i++)
#pragma unroll
        for (int r = 0; r < 8; r++) acc[i][r] = 0.f;

    for (int kc = 0; kc < 8; kc++) {
        // ---- A fragment: 16 tokens x 32 K of x, fp32 -> bf16 ----
        const int tok = t0 + lh;
        const float* xr = x + (size_t)tok * CCH + kc * 32 + base;
        float4 q0 = *(const float4*)(xr);
        float4 q1 = *(const float4*)(xr + 4);
        float4 q2 = *(const float4*)(xr + 16);
        float4 q3 = *(const float4*)(xr + 20);
        float tl[16] = { q0.x,q0.y,q0.z,q0.w, q1.x,q1.y,q1.z,q1.w,
                         q2.x,q2.y,q2.z,q2.w, q3.x,q3.y,q3.z,q3.w };
        Frag16 a;
#pragma unroll
        for (int j = 0; j < 16; j++) a.us[j] = f2bf(tl[j]);

#pragma unroll
        for (int i = 0; i < 5; i++) {
            const int tileId = wave + i * 4;
            const unsigned short* WT =
                (tileId < 2) ? wfT : ((tileId < 4) ? wgT : whT);
            const int col0 = (tileId < 2) ? tileId * 16
                           : (tileId < 4) ? (tileId - 2) * 16
                                          : (tileId - 4) * 16;
            const unsigned short* p = WT + (size_t)(col0 + lh) * 256 + kc * 32 + base;
            Frag16 b = load_frag(p);
            acc[i] = wmma_bf16(a, b, acc[i]);
        }
    }

    // ---- bias (+ log2e for g) + store ----
#pragma unroll
    for (int i = 0; i < 5; i++) {
        const int tileId = wave + i * 4;
        if (tileId < 4) {
            const bool isF = (tileId < 2);
            const float* bias = isF ? bfv : bgv;
            unsigned short* dst = isF ? fo : go;
            const float sc = isF ? 1.0f : LOG2E;   // exp2-domain scores
            const int col = (tileId & 1) * 16 + lh;
            const float bv = bias[col];
#pragma unroll
            for (int r = 0; r < 8; r++)
                dst[(size_t)(t0 + r + 8 * hh) * CK + col] =
                    f2bf((acc[i][r] + bv) * sc);
        } else {
            const int c = (tileId - 4) * 16 + lh;
            const float bv = bhv[c];
#pragma unroll
            for (int r = 0; r < 8; r++)
                hTo[(size_t)bb * (CCH * NTOK) + (size_t)c * NTOK
                    + n0 + r + 8 * hh] = f2bf(acc[i][r] + bv);
        }
    }
}

// ---------------------------------------------------------------------------
// Kernel 2: flash attention. Block = 256 thr = 8 waves; wave = (qt, cg):
// qt in {0,1} -> 16-query tile, cg in 0..3 -> 64 output channels.
// Per 64-key chunk: S (4 WMMAs), exp2-domain online softmax (rescale only
// when the running max moves, voted wave-uniformly), P through per-wave LDS
// tile, row-sums via 2 WMMAs against ones, O += P V (8 WMMAs).
// ---------------------------------------------------------------------------
__global__ __launch_bounds__(256) void attn_kernel(
    const unsigned short* __restrict__ ws, const float* __restrict__ x,
    const float* __restrict__ gammaP, float* __restrict__ y)
{
    __shared__ __align__(16) unsigned short P[8][1024];  // per-wave 16x64 bf16

    const unsigned short* g  = ws + OFF_G;
    const unsigned short* f  = ws + OFF_F;
    const unsigned short* hT = ws + OFF_HT;

    const int tid = threadIdx.x, wave = tid >> 5, lane = tid & 31;
    const int lh = lane & 15, hh = lane >> 4, base = hh * 8;
    const int b = blockIdx.x >> 7, qblk = blockIdx.x & 127;
    const int qt = wave >> 2, cg = wave & 3;
    const int q0 = qblk * 32 + qt * 16;

    const unsigned short* gq = g  + (size_t)b * (NTOK * CK);
    const unsigned short* fk = f  + (size_t)b * (NTOK * CK);
    const unsigned short* hv = hT + (size_t)b * (CCH * NTOK);

    // query A-fragment (persistent): rows q0..q0+15, K = all 32
    Frag16 ga = load_frag(gq + (size_t)(q0 + lh) * CK + base);

    // constant ones B-fragment: rowsum(P) = P @ ones via the matrix pipe
    Frag16 ones;
#pragma unroll
    for (int j = 0; j < 16; j++) ones.us[j] = 0x3F80;    // bf16 1.0

    float m[8];
    v8f acc[4], Lacc;
#pragma unroll
    for (int r = 0; r < 8; r++) { m[r] = -1e30f; Lacc[r] = 0.f; }
#pragma unroll
    for (int t = 0; t < 4; t++)
#pragma unroll
        for (int r = 0; r < 8; r++) acc[t][r] = 0.f;

    for (int k0 = 0; k0 < NTOK; k0 += 64) {
        // warm caches for the next chunk (global_prefetch_b8)
        if (k0 + 64 < NTOK) {
            __builtin_prefetch(fk + (size_t)(k0 + 64 + lh) * CK, 0, 0);
            __builtin_prefetch(hv + (size_t)((cg * 4) * 16 + lh) * NTOK + k0 + 64, 0, 0);
        }

        // ---- S = g f^T : B columns are key rows of f (contiguous) ----
        v8f z;
#pragma unroll
        for (int r = 0; r < 8; r++) z[r] = 0.f;
        v8f s[4];
#pragma unroll
        for (int j = 0; j < 4; j++) {
            Frag16 fb = load_frag(fk + (size_t)(k0 + j * 16 + lh) * CK + base);
            s[j] = wmma_bf16(ga, fb, z);
        }

        // ---- running-max update, skipped when max hasn't moved ----
        float mx4[8];
        int need = 0;
#pragma unroll
        for (int r = 0; r < 8; r++) {
            mx4[r] = fmaxf(fmaxf(s[0][r], s[1][r]), fmaxf(s[2][r], s[3][r]));
            need |= (mx4[r] > m[r]);
        }
        if (__any(need)) {           // wave-uniform; conservative per-row test
#pragma unroll
            for (int r = 0; r < 8; r++) {
                float mx = mx4[r];
                mx = fmaxf(mx, permxor(mx, XOR1_LO, XOR1_HI));
                mx = fmaxf(mx, permxor(mx, XOR2_LO, XOR2_HI));
                mx = fmaxf(mx, permxor(mx, XOR4_LO, XOR4_HI));
                mx = fmaxf(mx, permxor(mx, XOR8_LO, XOR8_HI));
                float mnew = fmaxf(m[r], mx);
                float sc = EXP2F(m[r] - mnew);
                m[r] = mnew;
                Lacc[r] *= sc;
#pragma unroll
                for (int t = 0; t < 4; t++) acc[t][r] *= sc;
            }
        }

        // ---- P = exp2(S - m), bf16, into per-wave LDS tile (D-layout) ----
#pragma unroll
        for (int r = 0; r < 8; r++) {
            const int row = r + 8 * hh;
            P[wave][row * 64 +      lh] = f2bf(EXP2F(s[0][r] - m[r]));
            P[wave][row * 64 + 16 + lh] = f2bf(EXP2F(s[1][r] - m[r]));
            P[wave][row * 64 + 32 + lh] = f2bf(EXP2F(s[2][r] - m[r]));
            P[wave][row * 64 + 48 + lh] = f2bf(EXP2F(s[3][r] - m[r]));
        }

        // P written in D-layout; re-read as A-layout (same wave -> dscnt only)
        asm volatile("s_wait_dscnt 0" ::: "memory");
        Frag16 pa0 = load_frag(&P[wave][lh * 64 + base]);
        Frag16 pa1 = load_frag(&P[wave][lh * 64 + 32 + base]);

        // row-sums on the matrix pipe (all 16 D columns hold the row sum)
        Lacc = wmma_bf16(pa0, ones, Lacc);
        Lacc = wmma_bf16(pa1, ones, Lacc);

        // ---- O += P V : B columns are channel rows of hT (contiguous) ----
#pragma unroll
        for (int t = 0; t < 4; t++) {
            const int c = (cg * 4 + t) * 16 + lh;
            Frag16 vb0 = load_frag(hv + (size_t)c * NTOK + k0 + base);
            Frag16 vb1 = load_frag(hv + (size_t)c * NTOK + k0 + 32 + base);
            acc[t] = wmma_bf16(pa0, vb0, acc[t]);
            acc[t] = wmma_bf16(pa1, vb1, acc[t]);
        }
        // make sure pa reads retire before next iteration overwrites P
        asm volatile("s_wait_dscnt 0" ::: "memory");
    }

    // ---- epilogue: y = gamma * (acc / l) + x ----
    const float gamma = gammaP[0];
    float inv[8];
#pragma unroll
    for (int r = 0; r < 8; r++) inv[r] = 1.0f / Lacc[r];
#pragma unroll
    for (int t = 0; t < 4; t++) {
#pragma unroll
        for (int r = 0; r < 8; r++) {
            const int row = q0 + r + 8 * hh;
            const int c = (cg * 4 + t) * 16 + lh;
            const size_t idx = ((size_t)b * NTOK + row) * CCH + c;
            y[idx] = gamma * acc[t][r] * inv[r] + x[idx];
        }
    }
}

// ---------------------------------------------------------------------------
extern "C" void kernel_launch(void* const* d_in, const int* in_sizes, int n_in,
                              void* d_out, int out_size, void* d_ws, size_t ws_size,
                              hipStream_t stream)
{
    (void)in_sizes; (void)n_in; (void)out_size; (void)ws_size;
    const float* x     = (const float*)d_in[0];
    const float* wf    = (const float*)d_in[1];
    const float* bfv   = (const float*)d_in[2];
    const float* wg    = (const float*)d_in[3];
    const float* bgv   = (const float*)d_in[4];
    const float* wh    = (const float*)d_in[5];
    const float* bhv   = (const float*)d_in[6];
    const float* gamma = (const float*)d_in[7];
    float* out = (float*)d_out;
    unsigned short* ws = (unsigned short*)d_ws;

    prep_kernel<<<320, 256, 0, stream>>>(wf, wg, wh, ws);
    proj_kernel<<<(NB * NTOK) / 16, 128, 0, stream>>>(x, ws, bfv, bgv, bhv, ws);
    attn_kernel<<<NB * (NTOK / 32), 256, 0, stream>>>(ws, x, gamma, out);
}
